// Block_45518063403599
// MI455X (gfx1250) — compile-verified, matrix operand order checked by reading, not run on hardware
//
#include <hip/hip_runtime.h>

// ---------------------------------------------------------------------------
// CDNA5 (gfx1250) transformer block forward.
//  * all matmuls: v_wmma_f32_16x16x32_bf16, fp32 accumulate
//  * 128x128x32 LDS tiles, double buffered
//  * tile staging via GLOBAL_LOAD_ASYNC_TO_LDS_B128 (ASYNCcnt-tracked DMA into
//    LDS, no VGPR round trip) -- every GEMM operand is laid out so its LDS
//    image is a contiguous [rows][32] copy (weights pre-transposed, V stored
//    transposed), so async b128 copies suffice; the only synchronous staging
//    path is the f32->bf16 convert of the attention probabilities.
// ---------------------------------------------------------------------------

typedef __attribute__((ext_vector_type(16))) __bf16 v16bf;
typedef __attribute__((ext_vector_type(8)))  float  v8f;

union FragU { uint4 u4[2]; v16bf v; };

__device__ __forceinline__ unsigned short f2bf(float f) {
  union { float f; unsigned u; } x; x.f = f;
  unsigned r = x.u + 0x7fffu + ((x.u >> 16) & 1u);   // round-to-nearest-even
  return (unsigned short)(r >> 16);
}

// Async DMA: global -> LDS, 16 bytes, tracked by ASYNCcnt.
// Flat LDS pointers carry the wave-relative LDS offset in their low 32 bits
// (ISA 10.2 aperture rules), which is exactly the VDST operand of the async op.
__device__ __forceinline__ void async_b128(void* lds_dst, const void* gsrc) {
  unsigned loff = (unsigned)(unsigned long long)lds_dst;
  unsigned long long ga = (unsigned long long)gsrc;
  asm volatile("global_load_async_to_lds_b128 %0, %1, off"
               :: "v"(loff), "v"(ga) : "memory");
}
__device__ __forceinline__ void wait_async() {
  asm volatile("s_wait_asynccnt 0x0" ::: "memory");
}

__device__ __forceinline__ v16bf ld_frag(const unsigned short (*tile)[32], int base, int lane) {
  const int r  = base + (lane & 15);
  const int kh = (lane >> 4) * 8;
  FragU u;
  u.u4[0] = *(const uint4*)&tile[r][kh];        // K = kh+0..7
  u.u4[1] = *(const uint4*)&tile[r][16 + kh];   // K = 16+kh+0..7
  return u.v;
}

// EPI: 0 = store bf16
//      1 = store f32 * scale
//      2 = store f32 + bias + residual
//      3 = store bf16 relu(acc + bias)
//      4 = store bf16 transposed (C[col][row], ldc = row count stride)
// B operand is ALWAYS [N][K] row-major bf16 ("NT").
template<bool A_F32, int EPI>
__global__ __launch_bounds__(256) void gemm_wmma(
    const void* __restrict__ Araw, const unsigned short* __restrict__ Braw,
    void* __restrict__ Craw, const float* __restrict__ bias,
    const float* __restrict__ resid,
    int M, int N, int K,
    long lda, long ldb, long ldc, long ldr,
    int zdiv,
    long sA1, long sA2, long sB1, long sB2,
    long sC1, long sC2, long sR1, long sR2,
    float scale)
{
  constexpr int BM = 128, BN = 128, KT = 32;
  __shared__ __align__(16) unsigned short As[2][BM][KT];
  __shared__ __align__(16) unsigned short Bs[2][BN][KT];

  const int tid  = threadIdx.x;
  const int lane = tid & 31;
  const int wave = tid >> 5;
  const int wm = (wave >> 2) * 64;   // wave row origin in block tile
  const int wn = (wave & 3) * 32;    // wave col origin in block tile

  const int z  = blockIdx.z;
  const int z1 = z / zdiv, z2 = z % zdiv;

  const float*          Af = (const float*)Araw          + (A_F32 ? ((long)z1 * sA1 + (long)z2 * sA2) : 0);
  const unsigned short* Au = (const unsigned short*)Araw + (A_F32 ? 0 : ((long)z1 * sA1 + (long)z2 * sA2));
  const unsigned short* Bp = Braw + (long)z1 * sB1 + (long)z2 * sB2;

  const int m0 = blockIdx.y * BM;
  const int n0 = blockIdx.x * BN;
  const int nk = K / KT;             // K is always a multiple of 32 here

  auto stageA = [&](int buf, int kt) {
    const int k0 = kt * KT;
    #pragma unroll
    for (int it = 0; it < 2; ++it) {
      int idx = tid + it * 256;      // 512 slots = 128 rows x 4 segs of 8 halves
      int row = idx >> 2;
      int seg = idx & 3;
      unsigned short* dst = &As[buf][row][seg * 8];
      long aoff = (long)(m0 + row) * lda + k0 + seg * 8;
      if constexpr (A_F32) {
        const float* src = Af + aoff;
        float4 f0 = *(const float4*)src;
        float4 f1 = *(const float4*)(src + 4);
        dst[0] = f2bf(f0.x); dst[1] = f2bf(f0.y); dst[2] = f2bf(f0.z); dst[3] = f2bf(f0.w);
        dst[4] = f2bf(f1.x); dst[5] = f2bf(f1.y); dst[6] = f2bf(f1.z); dst[7] = f2bf(f1.w);
        if (seg == 0 && kt + 2 < nk) __builtin_prefetch(src + 2 * KT, 0, 1);
      } else {
        async_b128(dst, Au + aoff);
      }
    }
  };

  auto stageB = [&](int buf, int kt) {
    const int k0 = kt * KT;
    #pragma unroll
    for (int it = 0; it < 2; ++it) {
      int idx = tid + it * 256;
      int n   = idx >> 2;
      int seg = idx & 3;
      unsigned short* dst = &Bs[buf][n][seg * 8];
      int gn = n0 + n;
      if (gn < N) {
        async_b128(dst, Bp + (long)gn * ldb + k0 + seg * 8);
      } else {
        uint4 zz = {0u, 0u, 0u, 0u};
        *(uint4*)dst = zz;              // zero-pad tail (only attnV: N=64)
      }
    }
  };

  v8f acc[4][2];
  {
    v8f zero = {};
    #pragma unroll
    for (int i = 0; i < 4; ++i) { acc[i][0] = zero; acc[i][1] = zero; }
  }

  stageA(0, 0); stageB(0, 0);
  wait_async();
  __syncthreads();

  int buf = 0;
  for (int kt = 0; kt < nk; ++kt) {
    if (kt + 1 < nk) { stageA(buf ^ 1, kt + 1); stageB(buf ^ 1, kt + 1); }

    v16bf bf0 = ld_frag(Bs[buf], wn, lane);
    v16bf bf1 = ld_frag(Bs[buf], wn + 16, lane);
    #pragma unroll
    for (int mi = 0; mi < 4; ++mi) {
      v16bf af = ld_frag(As[buf], wm + mi * 16, lane);
      acc[mi][0] = __builtin_amdgcn_wmma_f32_16x16x32_bf16(
          false, af, false, bf0, (short)0, acc[mi][0], false, false);
      acc[mi][1] = __builtin_amdgcn_wmma_f32_16x16x32_bf16(
          false, af, false, bf1, (short)0, acc[mi][1], false, false);
    }
    wait_async();                      // async copies of buf^1 complete
    __syncthreads();
    buf ^= 1;
  }

  // Epilogue. C layout (ISA 7.12.2): lane 0-15 -> N=lane, rows M = r (+8 hi half)
  const long coff = (long)z1 * sC1 + (long)z2 * sC2;
  const float* Rp = resid ? (resid + (long)z1 * sR1 + (long)z2 * sR2) : nullptr;
  #pragma unroll
  for (int mi = 0; mi < 4; ++mi) {
    #pragma unroll
    for (int ni = 0; ni < 2; ++ni) {
      int col = n0 + wn + ni * 16 + (lane & 15);
      if (col >= N) continue;
      int rbase = m0 + wm + mi * 16 + (lane >> 4) * 8;
      #pragma unroll
      for (int r = 0; r < 8; ++r) {
        long row = rbase + r;
        float v = acc[mi][ni][r];
        if constexpr (EPI == 0) {
          ((unsigned short*)Craw)[coff + row * ldc + col] = f2bf(v);
        } else if constexpr (EPI == 1) {
          ((float*)Craw)[coff + row * ldc + col] = v * scale;
        } else if constexpr (EPI == 2) {
          ((float*)Craw)[coff + row * ldc + col] = v + bias[col] + Rp[row * ldr + col];
        } else if constexpr (EPI == 3) {
          float t = v + bias[col];
          ((unsigned short*)Craw)[coff + row * ldc + col] = f2bf(t > 0.f ? t : 0.f);
        } else {
          ((unsigned short*)Craw)[coff + (long)col * ldc + row] = f2bf(v);
        }
      }
    }
  }
}

// -------------------------- elementwise kernels ----------------------------

__global__ void ln_bf16_kernel(const float* __restrict__ x,
                               const float* __restrict__ g,
                               const float* __restrict__ b,
                               unsigned short* __restrict__ out, int Dd) {
  __shared__ float s1[256], s2[256];
  const long row = blockIdx.x;
  const float* xr = x + row * Dd;
  const int tid = threadIdx.x;
  float sum = 0.f, sq = 0.f;
  for (int i = tid; i < Dd; i += 256) { float v = xr[i]; sum += v; sq += v * v; }
  s1[tid] = sum; s2[tid] = sq; __syncthreads();
  for (int st = 128; st > 0; st >>= 1) {
    if (tid < st) { s1[tid] += s1[tid + st]; s2[tid] += s2[tid + st]; }
    __syncthreads();
  }
  const float mu = s1[0] / Dd;
  const float var = s2[0] / Dd - mu * mu;
  const float rs = rsqrtf(var + 1e-5f);
  for (int i = tid; i < Dd; i += 256)
    out[row * Dd + i] = f2bf((xr[i] - mu) * rs * g[i] + b[i]);
}

__global__ void softmax_rows(float* __restrict__ p, int Tlen) {
  __shared__ float s[256];
  float* row = p + (long)blockIdx.x * Tlen;
  const int tid = threadIdx.x;
  float m = -3.4e38f;
  for (int i = tid; i < Tlen; i += 256) m = fmaxf(m, row[i]);
  s[tid] = m; __syncthreads();
  for (int st = 128; st > 0; st >>= 1) {
    if (tid < st) s[tid] = fmaxf(s[tid], s[tid + st]);
    __syncthreads();
  }
  const float M = s[0];
  __syncthreads();
  float sum = 0.f;
  for (int i = tid; i < Tlen; i += 256) {
    float e = __expf(row[i] - M);
    row[i] = e; sum += e;
  }
  s[tid] = sum; __syncthreads();
  for (int st = 128; st > 0; st >>= 1) {
    if (tid < st) s[tid] += s[tid + st];
    __syncthreads();
  }
  const float inv = 1.f / s[0];
  for (int i = tid; i < Tlen; i += 256) row[i] *= inv;
}

// [K][N] f32 -> [N][K] bf16 (transposed weight for NT GEMM staging)
__global__ void cvt_t_bf16(const float* __restrict__ in,
                           unsigned short* __restrict__ out, int Kd, int Nd) {
  long i = (long)blockIdx.x * blockDim.x + threadIdx.x;
  long n = (long)Kd * Nd;
  if (i >= n) return;
  int kk = (int)(i / Nd);
  int nn = (int)(i % Nd);
  out[(long)nn * Kd + kk] = f2bf(in[i]);
}

// [H, D, HD] f32 -> [H*HD][D] bf16 (head-concatenated, transposed projection)
__global__ void cvt_qkv_t_bf16(const float* __restrict__ in,
                               unsigned short* __restrict__ out,
                               int H_, int D_, int HD_) {
  long i = (long)blockIdx.x * blockDim.x + threadIdx.x;
  long n = (long)H_ * D_ * HD_;
  if (i >= n) return;
  int e = (int)(i % HD_);
  long t = i / HD_;
  int d = (int)(t % D_);
  int h = (int)(t / D_);
  out[((long)h * HD_ + e) * D_ + d] = f2bf(in[i]);
}

// ------------------------------- launcher ----------------------------------

extern "C" void kernel_launch(void* const* d_in, const int* in_sizes, int n_in,
                              void* d_out, int out_size, void* d_ws, size_t ws_size,
                              hipStream_t stream) {
  (void)in_sizes; (void)n_in; (void)out_size; (void)ws_size;
  constexpr int Bb = 4, T = 2048, Dm = 1024, H = 16, HD = 64, Dff = 4096;
  constexpr long MBT = (long)Bb * T;                 // 8192 token rows

  const float* x   = (const float*)d_in[0];
  const float* g1  = (const float*)d_in[1];
  const float* bt1 = (const float*)d_in[2];
  const float* g2  = (const float*)d_in[3];
  const float* bt2 = (const float*)d_in[4];
  const float* Wq  = (const float*)d_in[5];
  const float* Wk  = (const float*)d_in[6];
  const float* Wv  = (const float*)d_in[7];
  const float* Wp  = (const float*)d_in[8];
  const float* bp  = (const float*)d_in[9];
  const float* W1  = (const float*)d_in[10];
  const float* b1  = (const float*)d_in[11];
  const float* W2  = (const float*)d_in[12];
  const float* b2  = (const float*)d_in[13];

  float* xout = (float*)d_out;                       // [B,T,D] f32
  float* attn = xout + MBT * Dm;                     // [H,B,T,T] f32

  char* wsp = (char*)d_ws;
  size_t off = 0;
  auto alloc = [&](size_t bytes) -> void* {
    void* p = wsp + off;
    off = (off + bytes + 255) & ~(size_t)255;
    return p;
  };
  unsigned short* wqb  = (unsigned short*)alloc((size_t)Dm * Dm * 2);   // [D][D]  (W^T)
  unsigned short* wkb  = (unsigned short*)alloc((size_t)Dm * Dm * 2);
  unsigned short* wvb  = (unsigned short*)alloc((size_t)Dm * Dm * 2);
  unsigned short* wpb  = (unsigned short*)alloc((size_t)Dm * Dm * 2);
  unsigned short* w1b  = (unsigned short*)alloc((size_t)Dm * Dff * 2);  // [Dff][Dm]
  unsigned short* w2b  = (unsigned short*)alloc((size_t)Dff * Dm * 2);  // [Dm][Dff]
  unsigned short* hln  = (unsigned short*)alloc((size_t)MBT * Dm * 2);
  unsigned short* qcat = (unsigned short*)alloc((size_t)MBT * Dm * 2);  // [B*T][D]
  unsigned short* kcat = (unsigned short*)alloc((size_t)MBT * Dm * 2);  // [B*T][D]
  unsigned short* vT   = (unsigned short*)alloc((size_t)MBT * Dm * 2);  // [B][H][HD][T]
  unsigned short* ocat = (unsigned short*)alloc((size_t)MBT * Dm * 2);  // [B*T][D]
  unsigned short* h2   = (unsigned short*)alloc((size_t)MBT * Dm * 2);
  unsigned short* ff1  = (unsigned short*)alloc((size_t)MBT * Dff * 2);

  const dim3 blk(256);

  // 1) weights -> transposed bf16 ([N][K] for NT staging)
  {
    long nq = (long)H * Dm * HD;
    dim3 g((unsigned)((nq + 255) / 256));
    cvt_qkv_t_bf16<<<g, blk, 0, stream>>>(Wq, wqb, H, Dm, HD);
    cvt_qkv_t_bf16<<<g, blk, 0, stream>>>(Wk, wkb, H, Dm, HD);
    cvt_qkv_t_bf16<<<g, blk, 0, stream>>>(Wv, wvb, H, Dm, HD);
    long np = (long)Dm * Dm;
    cvt_t_bf16<<<dim3((unsigned)((np + 255) / 256)), blk, 0, stream>>>(Wp, wpb, Dm, Dm);
    long n1 = (long)Dm * Dff;
    cvt_t_bf16<<<dim3((unsigned)((n1 + 255) / 256)), blk, 0, stream>>>(W1, w1b, Dm, Dff);
    cvt_t_bf16<<<dim3((unsigned)((n1 + 255) / 256)), blk, 0, stream>>>(W2, w2b, Dff, Dm);
  }

  // 2) LN1 -> bf16
  ln_bf16_kernel<<<dim3((unsigned)MBT), blk, 0, stream>>>(x, g1, bt1, hln, Dm);

  // 3) Q/K projections: [8192,1024] = hln x Wt, bf16 row-major out
  {
    dim3 g(Dm / 128, (unsigned)(MBT / 128), 1);
    gemm_wmma<false, 0><<<g, blk, 0, stream>>>(
        hln, wqb, qcat, nullptr, nullptr, (int)MBT, Dm, Dm,
        Dm, Dm, Dm, 0, 1, 0, 0, 0, 0, 0, 0, 0, 0, 1.f);
    gemm_wmma<false, 0><<<g, blk, 0, stream>>>(
        hln, wkb, kcat, nullptr, nullptr, (int)MBT, Dm, Dm,
        Dm, Dm, Dm, 0, 1, 0, 0, 0, 0, 0, 0, 0, 0, 1.f);
  }

  // 3b) V projection, stored TRANSPOSED per batch: vT[b][h*HD+e][t]
  gemm_wmma<false, 4><<<dim3(Dm / 128, T / 128, Bb), blk, 0, stream>>>(
      hln, wvb, vT, nullptr, nullptr, T, Dm, Dm,
      Dm, Dm, T, 0, 1,
      (long)T * Dm, 0,                   // A: batch b rows of hln
      0, 0,
      (long)Dm * T, 0,                   // C: batch b slab of vT
      0, 0, 1.f);

  // 4) scores = (Q K^T) / sqrt(HD), f32 straight into d_out attn [H,B,T,T].
  //    z enumerates (b,h): z1=b, z2=h.
  gemm_wmma<false, 1><<<dim3(T / 128, T / 128, Bb * H), blk, 0, stream>>>(
      qcat, kcat, attn, nullptr, nullptr, T, T, HD,
      Dm, Dm, T, 0, H,
      (long)T * Dm, (long)HD,            // A offsets: b*T*D + h*HD
      (long)T * Dm, (long)HD,            // B offsets: same view of kcat
      (long)T * T, (long)Bb * T * T,     // C offsets: (h*B + b)*T*T
      0, 0, 0.125f);

  // 5) softmax in place over attn rows
  softmax_rows<<<dim3((unsigned)((long)Bb * H * T)), blk, 0, stream>>>(attn, T);

  // 6) O = attn x V   (A f32 converted during staging; B = vT is NT-native)
  gemm_wmma<true, 0><<<dim3(1, T / 128, Bb * H), blk, 0, stream>>>(
      attn, vT, ocat, nullptr, nullptr, T, HD, T,
      T, T, Dm, 0, H,
      (long)T * T, (long)Bb * T * T,     // A offsets into [H,B,T,T]
      (long)Dm * T, (long)HD * T,        // B offsets into vT [B][H][HD][T]
      (long)T * Dm, (long)HD,            // C offsets into ocat [B*T][D]
      0, 0, 1.f);

  // 7) x1 = x + O Wp + bp  -> d_out x section (f32)
  gemm_wmma<false, 2><<<dim3(Dm / 128, (unsigned)(MBT / 128), 1), blk, 0, stream>>>(
      ocat, wpb, xout, bp, x, (int)MBT, Dm, Dm,
      Dm, Dm, Dm, Dm, 1, 0, 0, 0, 0, 0, 0, 0, 0, 1.f);

  // 8) LN2 -> bf16
  ln_bf16_kernel<<<dim3((unsigned)MBT), blk, 0, stream>>>(xout, g2, bt2, h2, Dm);

  // 9) ff1 = relu(h2 W1 + b1), bf16
  gemm_wmma<false, 3><<<dim3(Dff / 128, (unsigned)(MBT / 128), 1), blk, 0, stream>>>(
      h2, w1b, ff1, b1, nullptr, (int)MBT, Dff, Dm,
      Dm, Dm, Dff, 0, 1, 0, 0, 0, 0, 0, 0, 0, 0, 1.f);

  // 10) x_out = x1 + ff1 W2 + b2, in-place on d_out x section
  gemm_wmma<false, 2><<<dim3(Dm / 128, (unsigned)(MBT / 128), 1), blk, 0, stream>>>(
      ff1, w2b, xout, b2, xout, (int)MBT, Dm, Dff,
      Dff, Dff, Dm, Dm, 1, 0, 0, 0, 0, 0, 0, 0, 0, 1.f);
}
